// PhiMoeSparseMoeBlock_57354993271388
// MI455X (gfx1250) — matrix operand order repaired and checked by reading, not compiled, not run on hardware
//
#include <hip/hip_runtime.h>
#include <hip/hip_bf16.h>

// Problem constants (from reference): B=4, S=2048 -> T=8192 tokens
#define T_TOK 8192
#define H_DIM 2048
#define F_DIM 4096
#define N_EXP 8

typedef __attribute__((ext_vector_type(16))) __bf16 v16bf;
typedef __attribute__((ext_vector_type(8)))  __bf16 v8bf;
typedef __attribute__((ext_vector_type(8)))  float  v8f;
typedef __attribute__((ext_vector_type(4)))  unsigned v4u;

// ---------------------------------------------------------------- helpers
__device__ __forceinline__ unsigned pk2(float a, float b) {
  unsigned short ua = __builtin_bit_cast(unsigned short, (__bf16)a);
  unsigned short ub = __builtin_bit_cast(unsigned short, (__bf16)b);
  return (unsigned)ua | ((unsigned)ub << 16);
}
__device__ __forceinline__ unsigned short f2bf(float a) {
  return __builtin_bit_cast(unsigned short, (__bf16)a);
}
__device__ __forceinline__ v8f v8f_zero() {
  v8f z;
#pragma unroll
  for (int i = 0; i < 8; ++i) z[i] = 0.f;
  return z;
}

// CDNA5 async copy: 16B global -> LDS, tracked by ASYNCcnt (ISA 15.18.3 op 98)
__device__ __forceinline__ void async_ld16(unsigned lds_byte_off, const void* g) {
  asm volatile("global_load_async_to_lds_b128 %0, %1, off"
               :: "v"(lds_byte_off), "v"(g) : "memory");
}
__device__ __forceinline__ void wait_async0() {
  asm volatile("s_wait_asynccnt 0x0" ::: "memory");
}

// A fragment: two contiguous 16B runs of a
// row-major [row][32*bf16] tile: lane<16 -> K0..15, lane>=16 -> K16..31 (matches B)
__device__ __forceinline__ v16bf load_a_frag(const unsigned short* tile_row0, int lane) {
  const unsigned short* p = tile_row0 + (lane & 15) * 32 + (lane >> 4) * 16;
  v8bf lo = *(const v8bf*)p;
  v8bf hi = *(const v8bf*)(p + 8);
  return __builtin_shufflevector(lo, hi, 0, 1, 2, 3, 4, 5, 6, 7,
                                 8, 9, 10, 11, 12, 13, 14, 15);
}
// B fragment: k = 16h + s (same order as A), i.e. one contiguous 32B run of a
// column-major [col][32*bf16] tile: lane<16 -> col K0..15, lane>=16 -> col K16..31
__device__ __forceinline__ v16bf load_b_frag(const unsigned short* tile_col0, int lane) {
  return *(const v16bf*)(tile_col0 + (lane & 15) * 32 + (lane >> 4) * 16);
}

// ---------------------------------------------------------------- kernel 0: zero out + counts
__global__ void zero_kernel(float* __restrict__ p, long n, int* __restrict__ counts) {
  long i = (long)blockIdx.x * blockDim.x + threadIdx.x;
  long stride = (long)gridDim.x * blockDim.x;
  for (; i < n; i += stride) p[i] = 0.f;
  if (blockIdx.x == 0 && threadIdx.x < N_EXP) counts[threadIdx.x] = 0;
}

// ---------------------------------------------------------------- kernel 1: router (one wave32 per token)
__global__ void router_kernel(const float* __restrict__ x, const float* __restrict__ gate,
                              float* __restrict__ logits, int* __restrict__ counts,
                              int* __restrict__ tokens, float* __restrict__ wts) {
  const int wave = threadIdx.x >> 5, lane = threadIdx.x & 31;
  const int t = blockIdx.x * 8 + wave;
  float p[N_EXP];
#pragma unroll
  for (int e = 0; e < N_EXP; ++e) p[e] = 0.f;
  const float* xr = x + (size_t)t * H_DIM;
  for (int j = 0; j < H_DIM / 32; ++j) {
    int h = lane + 32 * j;
    float xv = xr[h];
    const float* g = gate + (size_t)h * N_EXP;
#pragma unroll
    for (int e = 0; e < N_EXP; ++e) p[e] = fmaf(xv, g[e], p[e]);
  }
#pragma unroll
  for (int off = 16; off >= 1; off >>= 1) {
#pragma unroll
    for (int e = 0; e < N_EXP; ++e) p[e] += __shfl_xor(p[e], off, 32);
  }
  if (lane == 0) {
#pragma unroll
    for (int e = 0; e < N_EXP; ++e) logits[(size_t)t * N_EXP + e] = p[e];
    int e1 = 0; float m1 = p[0];
#pragma unroll
    for (int e = 1; e < N_EXP; ++e) if (p[e] > m1) { m1 = p[e]; e1 = e; }
    int e2 = (e1 == 0) ? 1 : 0; float m2 = p[e2];
#pragma unroll
    for (int e = 0; e < N_EXP; ++e) if (e != e1 && p[e] > m2) { m2 = p[e]; e2 = e; }
    float z = __expf(m2 - m1);          // softmax over two selected logits
    float inv = 1.f / (1.f + z);
    int i1 = atomicAdd(&counts[e1], 1);
    tokens[e1 * T_TOK + i1] = t; wts[e1 * T_TOK + i1] = inv;
    int i2 = atomicAdd(&counts[e2], 1);
    tokens[e2 * T_TOK + i2] = t; wts[e2 * T_TOK + i2] = z * inv;
  }
}

// ---------------------------------------------------------------- kernel 2: prefix of counts
__global__ void prefix_kernel(const int* __restrict__ counts, int* __restrict__ base) {
  if (threadIdx.x == 0) {
    int s = 0;
    for (int e = 0; e < N_EXP; ++e) { base[e] = s; s += counts[e]; }
  }
}

// ---------------------------------------------------------------- kernel 3: gathered GEMM1 -> silu(x@w1)*(x@w3) -> bf16 inter
// Tile M=128 tokens x N=64 F-cols, K stepped by 32 over H. 8 waves, wave w owns
// row-block w, 4 col-blocks x {w1,w3} = 8 WMMAs/wave/k-step. Register-pipelined
// global loads; linear LDS tiles (A row-major, B col-major) staged with b128 stores.
__global__ void __launch_bounds__(256)
expert_gateup_kernel(const float* __restrict__ x,
                     const float* __restrict__ w1,
                     const float* __restrict__ w3,
                     const int* __restrict__ tokens,
                     const int* __restrict__ counts,
                     const int* __restrict__ base,
                     unsigned short* __restrict__ inter) {
  const int e = blockIdx.z;
  const int cnt = counts[e];
  const int tm = blockIdx.y;
  if (tm * 128 >= cnt) return;
  const int fc = blockIdx.x;

  __shared__ __align__(16) unsigned short As[128 * 32];   // row-major  [row][k]
  __shared__ __align__(16) unsigned short B1s[64 * 32];   // col-major  [col][k]
  __shared__ __align__(16) unsigned short B3s[64 * 32];

  const int tid = threadIdx.x, wave = tid >> 5, lane = tid & 31;

  // A staging: row am, 16 consecutive k (64B contiguous global read)
  const int am = tid >> 1, ahalf = tid & 1;
  int arow = tm * 128 + am; if (arow >= cnt) arow = cnt - 1;
  const float* aptr = x + (size_t)tokens[e * T_TOK + arow] * H_DIM + ahalf * 16;
  // B staging: column n, 8 consecutive k (column walk, lane-coalesced over n)
  const int bn = tid & 63, bks = tid >> 6;
  const size_t bbase = (size_t)e * H_DIM * F_DIM + (size_t)(bks * 8) * F_DIM + (size_t)fc * 64 + bn;
  const float* w1p = w1 + bbase;
  const float* w3p = w3 + bbase;

  float af[16], b1f[8], b3f[8];
#pragma unroll
  for (int j = 0; j < 16; ++j) af[j] = aptr[j];
#pragma unroll
  for (int j = 0; j < 8; ++j) { b1f[j] = w1p[(size_t)j * F_DIM]; b3f[j] = w3p[(size_t)j * F_DIM]; }

  v8f acc1[4], acc3[4];
#pragma unroll
  for (int c = 0; c < 4; ++c) { acc1[c] = v8f_zero(); acc3[c] = v8f_zero(); }

  for (int kc = 0; kc < H_DIM / 32; ++kc) {
    __syncthreads();
    // pack + stage (2x b128 for A, 1x b128 per B matrix)
    v4u aw0 = { pk2(af[0], af[1]),  pk2(af[2], af[3]),  pk2(af[4], af[5]),  pk2(af[6], af[7]) };
    v4u aw1 = { pk2(af[8], af[9]),  pk2(af[10], af[11]), pk2(af[12], af[13]), pk2(af[14], af[15]) };
    *(v4u*)&As[am * 32 + ahalf * 16]     = aw0;
    *(v4u*)&As[am * 32 + ahalf * 16 + 8] = aw1;
    v4u bw1 = { pk2(b1f[0], b1f[1]), pk2(b1f[2], b1f[3]), pk2(b1f[4], b1f[5]), pk2(b1f[6], b1f[7]) };
    *(v4u*)&B1s[bn * 32 + bks * 8] = bw1;
    v4u bw3 = { pk2(b3f[0], b3f[1]), pk2(b3f[2], b3f[3]), pk2(b3f[4], b3f[5]), pk2(b3f[6], b3f[7]) };
    *(v4u*)&B3s[bn * 32 + bks * 8] = bw3;
    // pipeline next k-step's global loads (latency hides under WMMA phase)
    if (kc + 1 < H_DIM / 32) {
      const float* ap = aptr + (size_t)(kc + 1) * 32;
#pragma unroll
      for (int j = 0; j < 16; ++j) af[j] = ap[j];
      const float* bp1 = w1p + (size_t)(kc + 1) * 32 * F_DIM;
      const float* bp3 = w3p + (size_t)(kc + 1) * 32 * F_DIM;
#pragma unroll
      for (int j = 0; j < 8; ++j) { b1f[j] = bp1[(size_t)j * F_DIM]; b3f[j] = bp3[(size_t)j * F_DIM]; }
    }
    __syncthreads();
    v16bf a = load_a_frag(&As[wave * 16 * 32], lane);
#pragma unroll
    for (int c = 0; c < 4; ++c) {
      v16bf b1 = load_b_frag(&B1s[c * 16 * 32], lane);
      acc1[c] = __builtin_amdgcn_wmma_f32_16x16x32_bf16(false, a, false, b1,
                                                        (short)0, acc1[c], false, false);
      v16bf b3 = load_b_frag(&B3s[c * 16 * 32], lane);
      acc3[c] = __builtin_amdgcn_wmma_f32_16x16x32_bf16(false, a, false, b3,
                                                        (short)0, acc3[c], false, false);
    }
  }

  // epilogue: silu(gate)*up -> bf16 intermediate (compact rows: base[e]+row)
  const int ibase = base[e];
  const int mrow0 = tm * 128 + wave * 16 + 8 * (lane >> 4);
  const int ncol = lane & 15;
#pragma unroll
  for (int c = 0; c < 4; ++c) {
#pragma unroll
    for (int r = 0; r < 8; ++r) {
      int row = mrow0 + r;
      if (row < cnt) {
        float g = acc1[c][r];
        float u = acc3[c][r];
        float h = (g / (1.f + __expf(-g))) * u;   // silu(g) * u
        inter[(size_t)(ibase + row) * F_DIM + fc * 64 + c * 16 + ncol] = f2bf(h);
      }
    }
  }
}

// ---------------------------------------------------------------- kernel 4: down proj + weighted scatter-add
// A tile (bf16 inter rows, contiguous) copied with global_load_async_to_lds_b128.
__global__ void __launch_bounds__(256)
expert_down_kernel(const unsigned short* __restrict__ inter,
                   const float* __restrict__ w2,
                   const int* __restrict__ tokens,
                   const float* __restrict__ wts,
                   const int* __restrict__ counts,
                   const int* __restrict__ base,
                   float* __restrict__ outF) {
  const int e = blockIdx.z;
  const int cnt = counts[e];
  const int tm = blockIdx.y;
  if (tm * 128 >= cnt) return;
  const int hc = blockIdx.x;

  __shared__ __align__(16) unsigned short As[128 * 32];   // row-major [row][k], bf16
  __shared__ __align__(16) unsigned short Bs[64 * 32];    // col-major [col][k]

  const int tid = threadIdx.x, wave = tid >> 5, lane = tid & 31;
  const int ibase = base[e];

  // A: async copies, thread owns 32B of one row
  const int am = tid >> 1, ahalf = tid & 1;
  int arow = tm * 128 + am; if (arow >= cnt) arow = cnt - 1;
  const unsigned short* agp = inter + (size_t)(ibase + arow) * F_DIM + ahalf * 16;
  const unsigned a_lds = (unsigned)(uintptr_t)&As[am * 32 + ahalf * 16];

  // B: column walk of w2
  const int bn = tid & 63, bks = tid >> 6;
  const float* w2p = w2 + (size_t)e * F_DIM * H_DIM + (size_t)(bks * 8) * H_DIM
                        + (size_t)hc * 64 + bn;

  float bf[8];
#pragma unroll
  for (int j = 0; j < 8; ++j) bf[j] = w2p[(size_t)j * H_DIM];

  v8f acc[4];
#pragma unroll
  for (int c = 0; c < 4; ++c) acc[c] = v8f_zero();

  for (int kc = 0; kc < F_DIM / 32; ++kc) {
    __syncthreads();
    // A tile: two 16B async global->LDS copies per thread (ASYNCcnt)
    const unsigned short* ag = agp + (size_t)kc * 32;
    async_ld16(a_lds, ag);
    async_ld16(a_lds + 16, ag + 8);
    // B tile: convert + stage
    v4u bw = { pk2(bf[0], bf[1]), pk2(bf[2], bf[3]), pk2(bf[4], bf[5]), pk2(bf[6], bf[7]) };
    *(v4u*)&Bs[bn * 32 + bks * 8] = bw;
    if (kc + 1 < F_DIM / 32) {
      const float* bp = w2p + (size_t)(kc + 1) * 32 * H_DIM;
#pragma unroll
      for (int j = 0; j < 8; ++j) bf[j] = bp[(size_t)j * H_DIM];
    }
    wait_async0();
    __syncthreads();
    v16bf a = load_a_frag(&As[wave * 16 * 32], lane);
#pragma unroll
    for (int c = 0; c < 4; ++c) {
      v16bf b = load_b_frag(&Bs[c * 16 * 32], lane);
      acc[c] = __builtin_amdgcn_wmma_f32_16x16x32_bf16(false, a, false, b,
                                                       (short)0, acc[c], false, false);
    }
  }

  const int ncol = lane & 15;
#pragma unroll
  for (int r = 0; r < 8; ++r) {
    int row = tm * 128 + wave * 16 + 8 * (lane >> 4) + r;
    if (row < cnt) {
      int tok = tokens[e * T_TOK + row];
      float wt = wts[e * T_TOK + row];
      float* dst = outF + (size_t)tok * H_DIM + hc * 64 + ncol;
#pragma unroll
      for (int c = 0; c < 4; ++c) atomicAdd(dst + c * 16, wt * acc[c][r]);
    }
  }
}

// ---------------------------------------------------------------- launch
extern "C" void kernel_launch(void* const* d_in, const int* in_sizes, int n_in,
                              void* d_out, int out_size, void* d_ws, size_t ws_size,
                              hipStream_t stream) {
  const float* x  = (const float*)d_in[0];   // [T, H]
  const float* gw = (const float*)d_in[1];   // [H, E]
  const float* w1 = (const float*)d_in[2];   // [E, H, F]
  const float* w3 = (const float*)d_in[3];   // [E, H, F]
  const float* w2 = (const float*)d_in[4];   // [E, F, H]

  float* outF   = (float*)d_out;                      // final [T, H]
  float* logits = outF + (size_t)T_TOK * H_DIM;       // router_logits [T, E]

  char* ws = (char*)d_ws;
  int*   counts = (int*)ws;                                       // 8 ints
  int*   base   = (int*)(ws + 64);                                // 8 ints
  int*   tokens = (int*)(ws + 128);                               // E*T ints (256 KB)
  float* wgt    = (float*)(ws + 128 + (size_t)N_EXP * T_TOK * 4); // E*T floats (256 KB)
  unsigned short* inter = (unsigned short*)(ws + (1u << 20));     // 2T x F bf16 (128 MiB)

  zero_kernel<<<2048, 256, 0, stream>>>(outF, (long)T_TOK * H_DIM, counts);
  router_kernel<<<T_TOK / 8, 256, 0, stream>>>(x, gw, logits, counts, tokens, wgt);
  prefix_kernel<<<1, 32, 0, stream>>>(counts, base);
  expert_gateup_kernel<<<dim3(F_DIM / 64, T_TOK / 128, N_EXP), 256, 0, stream>>>(
      x, w1, w3, tokens, counts, base, inter);
  expert_down_kernel<<<dim3(H_DIM / 64, T_TOK / 128, N_EXP), 256, 0, stream>>>(
      inter, w2, tokens, wgt, counts, base, outF);
}
